// DPMamba_40802189312123
// MI455X (gfx1250) — matrix operand order repaired and checked by reading, not compiled, not run on hardware
//
#include <hip/hip_runtime.h>
#include <hip/hip_bf16.h>

#define N64   64
#define DIN   128
#define NLAY  4
#define KSEG  250
#define SSEG  34
#define KERN  16
#define STRD  8
#define NB    2
#define TLEN  32000
#define LENC  3999      /* (32000-16)/8+1 */
#define KS    8500      /* KSEG*SSEG */

typedef _Float16 h16;
typedef __attribute__((ext_vector_type(16))) _Float16 v16h;
typedef __attribute__((ext_vector_type(8)))  _Float16 v8h;
typedef __attribute__((ext_vector_type(8)))  float    v8f;

static __device__ __forceinline__ float d_sigm(float x){
  // fast sigmoid: v_exp_f32 + v_rcp_f32 (no IEEE div chain)
  return __builtin_amdgcn_rcpf(1.0f + __expf(-x));
}
static __device__ __forceinline__ float d_silu(float x){ return x*d_sigm(x); }
static __device__ __forceinline__ float d_softplus(float x){ return (x>20.0f)?x:log1pf(__expf(x)); }

// ---- WMMA fragment loaders (wave32 layouts, ISA 7.12.2) ----
// A 16x32 f16: lane L holds row M=L&15; elem e -> K = (e&7) + 8*(L>=16) + 16*(e>=8)
static __device__ __forceinline__ v16h load_a(const h16* base, int ld, int k0){
  int lane = threadIdx.x & 31;
  const h16* p = base + (size_t)(lane & 15)*ld + k0 + ((lane >> 4) << 3);
  v8h lo = *(const v8h*)p;
  v8h hi = *(const v8h*)(p + 16);
  v16h a;
  #pragma unroll
  for (int i = 0; i < 8; ++i){ a[i] = lo[i]; a[i+8] = hi[i]; }
  return a;
}
// B 32x16 f16 from global weights stored [ncols][kd] (K innermost), branch-free:
// lane L holds col N=L&15; elem e -> K = k0 + (e&7) + 8*(L>=16) + 16*(e>=8)
static __device__ __forceinline__ v16h load_b(const h16* w, int kd, int col0, int k0){
  int lane = threadIdx.x & 31;
  const h16* p = w + (size_t)(col0 + (lane & 15))*kd + k0 + ((lane >> 4) << 3);
  v8h lo = *(const v8h*)p;
  v8h hi = *(const v8h*)(p + 16);
  v16h b;
  #pragma unroll
  for (int i = 0; i < 8; ++i){ b[i] = lo[i]; b[i+8] = hi[i]; }
  return b;
}

// ---- f32 -> f16 weight conversion ----
__global__ void wconv_kernel(const float* __restrict__ src, h16* __restrict__ dst, int n){
  int i = blockIdx.x*256 + threadIdx.x;
  if (i < n) dst[i] = (h16)src[i];
}

// ---- f32 -> f16 conversion of x_w with zero-padding 36 -> 48 rows ----
__global__ void wconv_pad48_kernel(const float* __restrict__ src, h16* __restrict__ dst){
  int i = blockIdx.x*256 + threadIdx.x;
  if (i >= 16*48*128) return;
  int k = i & 127; int j = (i >> 7) % 48; int w = i / (48*128);
  dst[i] = (j < 36) ? (h16)src[((size_t)w*36 + j)*128 + k] : (h16)0.0f;
}

// ---- encoder: strided conv1d + relu -> mixw[b][n][l] ----
__global__ void enc_kernel(const float* __restrict__ mix, const float* __restrict__ ew, float* __restrict__ mixw){
  int i = blockIdx.x*256 + threadIdx.x;
  if (i >= NB*N64*LENC) return;
  int l = i % LENC; int n = (i / LENC) % N64; int b = i / (LENC*N64);
  const float* mp = mix + (size_t)b*TLEN + (size_t)l*STRD;
  const float* wp = ew + n*KERN;
  float s = 0.f;
  #pragma unroll
  for (int k = 0; k < KERN; ++k) s += mp[k]*wp[k];
  mixw[i] = s > 0.f ? s : 0.f;
}

// ---- two-stage deterministic gLN statistics ----
__global__ void reduce_plain(const float* __restrict__ x, int perB, float* __restrict__ part){
  __shared__ float s1[256], s2[256];
  int b = blockIdx.x >> 6, blk = blockIdx.x & 63;
  const float* xb = x + (size_t)b*perB;
  float a = 0.f, q = 0.f;
  for (int i = blk*256 + threadIdx.x; i < perB; i += 64*256){ float v = xb[i]; a += v; q += v*v; }
  s1[threadIdx.x] = a; s2[threadIdx.x] = q; __syncthreads();
  for (int off = 128; off > 0; off >>= 1){
    if (threadIdx.x < off){ s1[threadIdx.x] += s1[threadIdx.x+off]; s2[threadIdx.x] += s2[threadIdx.x+off]; }
    __syncthreads();
  }
  if (threadIdx.x == 0){ part[blockIdx.x*2] = s1[0]; part[blockIdx.x*2+1] = s2[0]; }
}

__global__ void finalize_stats(const float* __restrict__ part, float* __restrict__ fin, float cnt, float eps){
  if (threadIdx.x == 0){
    int b = blockIdx.x;
    float s = 0.f, q = 0.f;
    for (int i = 0; i < 64; ++i){ s += part[(b*64+i)*2]; q += part[(b*64+i)*2+1]; }
    float m = s/cnt; float v = q/cnt - m*m;
    fin[b*2] = m; fin[b*2+1] = rsqrtf(v + eps);
  }
}

// ---- gLN3 normalize fused with bottleneck 64x64 ----
__global__ void bott_kernel(const float* __restrict__ mixw, const float* __restrict__ bw,
                            const float* __restrict__ g3w, const float* __restrict__ g3b,
                            const float* __restrict__ fin, float* __restrict__ out){
  int i = blockIdx.x*256 + threadIdx.x;
  if (i >= NB*N64*LENC) return;
  int l = i % LENC; int o = (i / LENC) % N64; int b = i / (LENC*N64);
  float m = fin[b*2], inv = fin[b*2+1];
  const float* col = mixw + (size_t)b*N64*LENC + l;
  float s = 0.f;
  for (int n = 0; n < 64; ++n){
    float hn = g3w[n]*(col[(size_t)n*LENC]-m)*inv + g3b[n];
    s += bw[o*64+n]*hn;
  }
  out[i] = s;
}

// ---- segmentation -> x4[b][n][k][s] ----
__global__ void seg_kernel(const float* __restrict__ h2, float* __restrict__ x4){
  int i = blockIdx.x*256 + threadIdx.x;
  if (i >= NB*N64*KS) return;
  int s = i % SSEG; int k = (i / SSEG) % KSEG; int n = (i / KS) % N64; int b = i / (KS*N64);
  int base = (s >> 1)*KSEG + (s & 1)*125 + k;
  float v = 0.f;
  if (base >= 125 && base < 125 + LENC) v = h2[((size_t)b*N64 + n)*LENC + base - 125];
  x4[i] = v;
}

// ---- per-token layernorm -> xn[seq][t][64] ----
__global__ void ln_kernel(const float* __restrict__ x4, const float* __restrict__ lw, const float* __restrict__ lb,
                          float* __restrict__ xn, int path, int T, int nseq){
  int i = blockIdx.x*256 + threadIdx.x;
  if (i >= nseq*T) return;
  int seq = i / T, t = i % T;
  int b, k, s;
  if (path == 0){ b = seq / SSEG; s = seq % SSEG; k = t; }
  else          { b = seq / KSEG; k = seq % KSEG; s = t; }
  size_t base = ((size_t)b*N64*KSEG + k)*SSEG + s; // +n*KS per channel
  float m = 0.f;
  for (int n = 0; n < 64; ++n) m += x4[base + (size_t)n*KS];
  m *= (1.0f/64.0f);
  float v = 0.f;
  for (int n = 0; n < 64; ++n){ float d = x4[base + (size_t)n*KS] - m; v += d*d; }
  v *= (1.0f/64.0f);
  float inv = rsqrtf(v + 1e-5f);
  float* o = xn + (size_t)i*64;
  for (int n = 0; n < 64; ++n) o[n] = lw[n]*(x4[base + (size_t)n*KS]-m)*inv + lb[n];
}

// ---- Mamba block: one workgroup per (sequence, direction) ----
__global__ __launch_bounds__(256) void mamba_kernel(
    const float* __restrict__ xn, float* __restrict__ fbuf,
    const h16* __restrict__ w_in, const h16* __restrict__ w_x, const h16* __restrict__ w_o,
    const float* __restrict__ conv_w, const float* __restrict__ conv_b,
    const float* __restrict__ dt_w, const float* __restrict__ dt_b,
    const float* __restrict__ Alog, const float* __restrict__ Dp,
    int li, int path, int T, int nseq)
{
  __shared__ __align__(16) h16   xn_h[16][64];
  __shared__ __align__(16) float xi_raw[16][128];   // reused as y after conv
  __shared__ __align__(16) h16   z_h[16][128];
  __shared__ __align__(16) float tail[3][128];
  __shared__ __align__(16) float xi_c[16][128];
  __shared__ __align__(16) h16   xi_h[16][128];     // reused as gated-y for out GEMM
  __shared__ __align__(16) float proj[16][48];
  __shared__ __align__(16) float delta_s[16][128];
  __shared__ __align__(16) float A_s[128][16];
  __shared__ __align__(16) float convw_s[128][4];
  __shared__ __align__(16) float dtw_s[128][4];
  __shared__ float dtb_s[128];
  __shared__ float Dps[128];
  __shared__ float convb_s[128];

  const int tid  = threadIdx.x;
  const int lane = tid & 31;
  // wave id as a proven-scalar value: tile-selection branches become s_cbranch
  const int wav  = __builtin_amdgcn_readfirstlane(tid >> 5);
  const int dir  = blockIdx.x & 1;
  const int seq  = blockIdx.x >> 1;
  const int widx = (li*2 + path)*2 + dir;

  const h16* Win = w_in + (size_t)widx*256*64;
  const h16* Wx  = w_x  + (size_t)widx*48*128;
  const h16* Wo  = w_o  + (size_t)widx*64*128;
  const float* Al  = Alog   + (size_t)widx*128*16;
  const float* cwp = conv_w + (size_t)widx*128*4;
  const float* dwp = dt_w   + (size_t)widx*128*4;
  const float* cbp = conv_b + (size_t)widx*128;
  const float* dbp = dt_b   + (size_t)widx*128;
  const float* Dpt = Dp     + (size_t)widx*128;

  for (int i = tid; i < 2048; i += 256) A_s[i>>4][i&15] = -__expf(Al[i]);
  for (int i = tid; i < 512; i += 256){ convw_s[i>>2][i&3] = cwp[i]; dtw_s[i>>2][i&3] = dwp[i]; }
  for (int i = tid; i < 128; i += 256){ dtb_s[i] = dbp[i]; Dps[i] = Dpt[i]; convb_s[i] = cbp[i]; }
  for (int i = tid; i < 384; i += 256) tail[i>>7][i&127] = 0.0f;

  float hreg[8];
  #pragma unroll
  for (int i = 0; i < 8; ++i) hreg[i] = 0.0f;

  const int dch = tid >> 1;
  const int sb  = (tid & 1) * 8;
  float* fout = fbuf + ((size_t)(dir*nseq + seq))*T*64;
  const int nch = (T + 15) >> 4;

  for (int c = 0; c < nch; ++c){
    const int t0 = c << 4;
    int ct = T - t0; if (ct > 16) ct = 16;
    const bool full = (ct == 16);
    __syncthreads();
    // 1. stage xn chunk (time-reversed for backward direction); uniform branches
    if (full){
      if (dir == 0){
        for (int i = tid; i < 1024; i += 256){
          int tt = i >> 6, n = i & 63;
          xn_h[tt][n] = (h16)xn[((size_t)seq*T + t0 + tt)*64 + n];
        }
      } else {
        for (int i = tid; i < 1024; i += 256){
          int tt = i >> 6, n = i & 63;
          xn_h[tt][n] = (h16)xn[((size_t)seq*T + (T-1-(t0+tt)))*64 + n];
        }
      }
    } else {
      for (int i = tid; i < 1024; i += 256){
        int tt = i >> 6, n = i & 63;
        int tg = t0 + tt;
        float v = 0.0f;
        if (tg < T){
          int tm = dir ? (T-1-tg) : tg;
          v = xn[((size_t)seq*T + tm)*64 + n];
        }
        xn_h[tt][n] = (h16)v;
      }
    }
    __syncthreads();
    // 2. in-projection: [16x64] @ [64x256]; two tiles per wave, scalar tile branch
    {
      // tile ti = wav (xi half, ti<8)
      int col0 = wav << 4;
      v8f acc;
      #pragma unroll
      for (int i = 0; i < 8; ++i) acc[i] = 0.0f;
      #pragma unroll
      for (int k0 = 0; k0 < 64; k0 += 32){
        v16h a = load_a(&xn_h[0][0], 64, k0);
        v16h b = load_b(Win, 64, col0, k0);
        acc = __builtin_amdgcn_wmma_f32_16x16x32_f16(false, a, false, b, (short)0, acc, false, false);
      }
      int rb  = (lane >> 4) * 8;
      int col = col0 + (lane & 15);
      #pragma unroll
      for (int j = 0; j < 8; ++j) xi_raw[rb+j][col] = acc[j];
    }
    {
      // tile ti = wav+8 (z half)
      int col0 = (wav + 8) << 4;
      v8f acc;
      #pragma unroll
      for (int i = 0; i < 8; ++i) acc[i] = 0.0f;
      #pragma unroll
      for (int k0 = 0; k0 < 64; k0 += 32){
        v16h a = load_a(&xn_h[0][0], 64, k0);
        v16h b = load_b(Win, 64, col0, k0);
        acc = __builtin_amdgcn_wmma_f32_16x16x32_f16(false, a, false, b, (short)0, acc, false, false);
      }
      int rb  = (lane >> 4) * 8;
      int col = col0 - 128 + (lane & 15);
      #pragma unroll
      for (int j = 0; j < 8; ++j) z_h[rb+j][col] = (h16)d_silu(acc[j]);
    }
    __syncthreads();
    // 3. depthwise causal conv (D_CONV=4) + SiLU
    for (int q = tid; q < 2048; q += 256){
      int tt = q >> 7, d = q & 127;
      float s = convb_s[d];
      #pragma unroll
      for (int j = 0; j < 4; ++j){
        int tp = tt - 3 + j;
        float v = (tp < 0) ? tail[3+tp][d] : xi_raw[tp][d];
        s += convw_s[d][j]*v;
      }
      float a = d_silu(s);
      xi_c[tt][d] = a;
      xi_h[tt][d] = (h16)a;
    }
    __syncthreads();
    if (full){
      for (int i = tid; i < 384; i += 256) tail[i>>7][i&127] = xi_raw[13 + (i>>7)][i&127];
    }
    __syncthreads();
    // 4. x-projection: [16x128] @ [128x48] (dt|B|C, padded; branch-free B loads)
    if (wav < 3){
      int col0 = wav << 4;
      v8f acc;
      #pragma unroll
      for (int i = 0; i < 8; ++i) acc[i] = 0.0f;
      #pragma unroll
      for (int k0 = 0; k0 < 128; k0 += 32){
        v16h a = load_a(&xi_h[0][0], 128, k0);
        v16h b = load_b(Wx, 128, col0, k0);
        acc = __builtin_amdgcn_wmma_f32_16x16x32_f16(false, a, false, b, (short)0, acc, false, false);
      }
      int col = col0 + (lane & 15);
      int rb  = (lane >> 4) * 8;
      #pragma unroll
      for (int j = 0; j < 8; ++j) proj[rb+j][col] = acc[j];
    }
    __syncthreads();
    // 5. delta = softplus(dt @ dtw^T + dtb)
    for (int q = tid; q < 2048; q += 256){
      int tt = q >> 7, d = q & 127;
      float s = dtb_s[d];
      #pragma unroll
      for (int r = 0; r < 4; ++r) s += proj[tt][r]*dtw_s[d][r];
      delta_s[tt][d] = d_softplus(s);
    }
    __syncthreads();
    // 6. selective scan (state in registers; xi_raw reused as y)
    float* yb = &xi_raw[0][0];
    for (int tt = 0; tt < ct; ++tt){
      float dtv = delta_s[tt][dch];
      float dx  = dtv * xi_c[tt][dch];
      float acc = 0.0f;
      #pragma unroll
      for (int i = 0; i < 8; ++i){
        int st = sb + i;
        float dA = __expf(dtv * A_s[dch][st]);
        hreg[i] = dA*hreg[i] + dx*proj[tt][4+st];
        acc += hreg[i]*proj[tt][20+st];
      }
      acc += __shfl_xor(acc, 1, 32);
      if ((tid & 1) == 0) yb[tt*128 + dch] = acc;
    }
    __syncthreads();
    // 7. gate: (y + xi*D) * silu(z) -> f16 (reuse xi_h)
    for (int q = tid; q < 2048; q += 256){
      int tt = q >> 7, d = q & 127;
      float v = (yb[tt*128+d] + xi_c[tt][d]*Dps[d]) * (float)z_h[tt][d];
      xi_h[tt][d] = (h16)v;
    }
    __syncthreads();
    // 8. out-projection: [16x128] @ [128x64]; uniform full/dir branches on store
    if (wav < 4){
      int col0 = wav << 4;
      v8f acc;
      #pragma unroll
      for (int i = 0; i < 8; ++i) acc[i] = 0.0f;
      #pragma unroll
      for (int k0 = 0; k0 < 128; k0 += 32){
        v16h a = load_a(&xi_h[0][0], 128, k0);
        v16h b = load_b(Wo, 128, col0, k0);
        acc = __builtin_amdgcn_wmma_f32_16x16x32_f16(false, a, false, b, (short)0, acc, false, false);
      }
      int col = col0 + (lane & 15);
      int rb  = (lane >> 4) * 8;
      if (full){
        if (dir == 0){
          float* p = fout + (size_t)(t0+rb)*64 + col;
          #pragma unroll
          for (int j = 0; j < 8; ++j) p[(size_t)j*64] = acc[j];
        } else {
          float* p = fout + (size_t)(T-1-(t0+rb))*64 + col;
          #pragma unroll
          for (int j = 0; j < 8; ++j) p[-(size_t)j*64] = acc[j];
        }
      } else {
        #pragma unroll
        for (int j = 0; j < 8; ++j){
          int tg = t0 + rb + j;
          if (tg < T){
            int tm = dir ? (T-1-tg) : tg;
            fout[(size_t)tm*64 + col] = acc[j];
          }
        }
      }
    }
  }
}

// ---- reduce over ir = x4 + fwd + bwd (for gLN4) ----
__global__ void reduce_ir(const float* __restrict__ x4, const float* __restrict__ fbuf,
                          float* __restrict__ part, int path, int T, int nseq){
  __shared__ float s1[256], s2[256];
  int b = blockIdx.x >> 6, blk = blockIdx.x & 63;
  const size_t fstride = (size_t)nseq*T*64;
  float a = 0.f, q = 0.f;
  for (int j = blk*256 + threadIdx.x; j < N64*KS; j += 64*256){
    int n = j / KS; int r = j % KS; int k = r / SSEG; int s = r % SSEG;
    size_t xi = ((size_t)(b*N64 + n)*KSEG + k)*SSEG + s;
    size_t fi = (path == 0) ? (((size_t)(b*SSEG + s))*T + k)*64 + n
                            : (((size_t)(b*KSEG + k))*T + s)*64 + n;
    float v = x4[xi] + fbuf[fi] + fbuf[fi + fstride];
    a += v; q += v*v;
  }
  s1[threadIdx.x] = a; s2[threadIdx.x] = q; __syncthreads();
  for (int off = 128; off > 0; off >>= 1){
    if (threadIdx.x < off){ s1[threadIdx.x] += s1[threadIdx.x+off]; s2[threadIdx.x] += s2[threadIdx.x+off]; }
    __syncthreads();
  }
  if (threadIdx.x == 0){ part[blockIdx.x*2] = s1[0]; part[blockIdx.x*2+1] = s2[0]; }
}

// ---- x4 = gLN4(x4 + f + bw) + x4 (in place) ----
__global__ void combine_kernel(float* __restrict__ x4, const float* __restrict__ fbuf,
                               const float* __restrict__ g4w, const float* __restrict__ g4b,
                               const float* __restrict__ fin, int path, int T, int nseq){
  int i = blockIdx.x*256 + threadIdx.x;
  if (i >= NB*N64*KS) return;
  int s = i % SSEG; int k = (i / SSEG) % KSEG; int n = (i / KS) % N64; int b = i / (KS*N64);
  const size_t fstride = (size_t)nseq*T*64;
  size_t fi = (path == 0) ? (((size_t)(b*SSEG + s))*T + k)*64 + n
                          : (((size_t)(b*KSEG + k))*T + s)*64 + n;
  float ir = x4[i] + fbuf[fi] + fbuf[fi + fstride];
  float m = fin[b*2], inv = fin[b*2+1];
  x4[i] = g4w[n]*(ir-m)*inv + g4b[n] + x4[i];
}

// ---- PReLU + 1x1 conv2d ----
__global__ void conv2d_kernel(const float* __restrict__ x4, const float* __restrict__ cw,
                              const float* __restrict__ cb, const float* __restrict__ pa,
                              float* __restrict__ out){
  int i = blockIdx.x*256 + threadIdx.x;
  if (i >= NB*N64*KS) return;
  int s = i % SSEG; int k = (i / SSEG) % KSEG; int o = (i / KS) % N64; int b = i / (KS*N64);
  float a = pa[0];
  float sum = cb[o];
  for (int n = 0; n < 64; ++n){
    float v = x4[((size_t)(b*N64 + n)*KSEG + k)*SSEG + s];
    v = (v >= 0.f) ? v : a*v;
    sum += cw[o*64+n]*v;
  }
  out[i] = sum;
}

// ---- overlap-add ----
__global__ void overadd_kernel(const float* __restrict__ b5, float* __restrict__ hoa){
  int i = blockIdx.x*256 + threadIdx.x;
  if (i >= NB*N64*LENC) return;
  int l = i % LENC; int n = (i / LENC) % N64; int b = i / (LENC*N64);
  int p1 = l + 125;
  int s1 = (p1 / KSEG)*2, k1 = p1 % KSEG;
  int s2 = (l / KSEG)*2 + 1, k2 = l % KSEG;
  size_t base = ((size_t)b*N64 + n)*KS;
  hoa[i] = b5[base + (size_t)k1*SSEG + s1] + b5[base + (size_t)k2*SSEG + s2];
}

// ---- tanh(out_w@h+b) * sigmoid(gate_w@h+b) ----
__global__ void mask1_kernel(const float* __restrict__ hoa, const float* __restrict__ ow, const float* __restrict__ ob,
                             const float* __restrict__ gw, const float* __restrict__ gb, float* __restrict__ m1){
  int i = blockIdx.x*256 + threadIdx.x;
  if (i >= NB*N64*LENC) return;
  int l = i % LENC; int o = (i / LENC) % N64; int b = i / (LENC*N64);
  const float* col = hoa + (size_t)b*N64*LENC + l;
  float s1 = ob[o], s2 = gb[o];
  for (int n = 0; n < 64; ++n){
    float v = col[(size_t)n*LENC];
    s1 += ow[o*64+n]*v; s2 += gw[o*64+n]*v;
  }
  m1[i] = tanhf(s1)*d_sigm(s2);
}

// ---- relu(end_w@m1) * mix_w ----
__global__ void mask2_kernel(const float* __restrict__ m1, const float* __restrict__ ew,
                             const float* __restrict__ mixw, float* __restrict__ mm){
  int i = blockIdx.x*256 + threadIdx.x;
  if (i >= NB*N64*LENC) return;
  int l = i % LENC; int o = (i / LENC) % N64; int b = i / (LENC*N64);
  const float* col = m1 + (size_t)b*N64*LENC + l;
  float s = 0.f;
  for (int n = 0; n < 64; ++n) s += ew[o*64+n]*col[(size_t)n*LENC];
  s = s > 0.f ? s : 0.f;
  mm[i] = s * mixw[i];
}

// ---- transposed-conv decoder (overlap-add over <=2 frames per sample) ----
__global__ void dec_kernel(const float* __restrict__ mm, const float* __restrict__ dw, float* __restrict__ est){
  int i = blockIdx.x*256 + threadIdx.x;
  if (i >= NB*TLEN) return;
  int t = i % TLEN; int b = i / TLEN;
  float s = 0.f;
  int lmin = t - 15; lmin = (lmin < 0) ? 0 : (lmin + 7) / 8;
  int lmax = t / 8;  if (lmax > LENC-1) lmax = LENC-1;
  for (int l = lmin; l <= lmax; ++l){
    int tap = t - 8*l;                       // in [0,15]
    const float* col = mm + (size_t)b*N64*LENC + l;
    for (int n = 0; n < 64; ++n) s += dw[n*KERN + tap]*col[(size_t)n*LENC];
  }
  est[i] = s;
}

extern "C" void kernel_launch(void* const* d_in, const int* in_sizes, int n_in,
                              void* d_out, int out_size, void* d_ws, size_t ws_size,
                              hipStream_t stream){
  (void)in_sizes; (void)n_in; (void)out_size; (void)ws_size;
  const float* mixture  = (const float*)d_in[0];
  const float* enc_w    = (const float*)d_in[1];
  const float* gln3_w   = (const float*)d_in[2];
  const float* gln3_b   = (const float*)d_in[3];
  const float* bott_w   = (const float*)d_in[4];
  const float* mm_in_w  = (const float*)d_in[5];
  const float* mm_conv_w= (const float*)d_in[6];
  const float* mm_conv_b= (const float*)d_in[7];
  const float* mm_x_w   = (const float*)d_in[8];
  const float* mm_dt_w  = (const float*)d_in[9];
  const float* mm_dt_b  = (const float*)d_in[10];
  const float* mm_Alog  = (const float*)d_in[11];
  const float* mm_D     = (const float*)d_in[12];
  const float* mm_out_w = (const float*)d_in[13];
  const float* ln_w     = (const float*)d_in[14];
  const float* ln_b     = (const float*)d_in[15];
  const float* gln4_w   = (const float*)d_in[16];
  const float* gln4_b   = (const float*)d_in[17];
  const float* prelu_a  = (const float*)d_in[18];
  const float* conv2d_w = (const float*)d_in[19];
  const float* conv2d_b = (const float*)d_in[20];
  const float* out_w    = (const float*)d_in[21];
  const float* out_b    = (const float*)d_in[22];
  const float* gate_w   = (const float*)d_in[23];
  const float* gate_b   = (const float*)d_in[24];
  const float* end_w    = (const float*)d_in[25];
  const float* dec_w    = (const float*)d_in[26];
  float* est = (float*)d_out;

  char* ws = (char*)d_ws;
  size_t off = 0;
  auto alloc = [&](size_t bytes)->char*{ char* p = ws + off; off += (bytes + 255) & ~(size_t)255; return p; };
  float* mixw  = (float*)alloc(sizeof(float)*NB*N64*LENC);
  float* hbott = (float*)alloc(sizeof(float)*NB*N64*LENC);
  float* x4    = (float*)alloc(sizeof(float)*NB*N64*KS);
  float* fbuf  = (float*)alloc(sizeof(float)*2*NB*KS*64);          // [2dir][nseq][T][64], same size both paths
  float* xnb   = (float*)alloc(sizeof(float)*NB*KS*64);
  float* buf5  = (float*)alloc(sizeof(float)*NB*N64*KS);
  float* hoa   = (float*)alloc(sizeof(float)*NB*N64*LENC);
  float* m1    = (float*)alloc(sizeof(float)*NB*N64*LENC);
  float* mmk   = (float*)alloc(sizeof(float)*NB*N64*LENC);
  float* stats = (float*)alloc(4096);
  h16* w_in16  = (h16*)alloc(sizeof(h16)*16*256*64);
  h16* w_x16   = (h16*)alloc(sizeof(h16)*16*48*128);   // 36 -> 48 zero-padded
  h16* w_o16   = (h16*)alloc(sizeof(h16)*16*64*128);
  float* fin  = stats;
  float* part = stats + 16;

  auto cdiv = [](int a, int b){ return (a + b - 1)/b; };

  wconv_kernel<<<cdiv(16*256*64,256),256,0,stream>>>(mm_in_w,  w_in16, 16*256*64);
  wconv_pad48_kernel<<<cdiv(16*48*128,256),256,0,stream>>>(mm_x_w, w_x16);
  wconv_kernel<<<cdiv(16*64*128,256),256,0,stream>>>(mm_out_w, w_o16,  16*64*128);

  const int nL  = cdiv(NB*N64*LENC, 256);
  const int nX4 = cdiv(NB*N64*KS, 256);

  enc_kernel<<<nL,256,0,stream>>>(mixture, enc_w, mixw);
  reduce_plain<<<NB*64,256,0,stream>>>(mixw, N64*LENC, part);
  finalize_stats<<<NB,32,0,stream>>>(part, fin, (float)(N64*LENC), 1e-8f);
  bott_kernel<<<nL,256,0,stream>>>(mixw, bott_w, gln3_w, gln3_b, fin, hbott);
  seg_kernel<<<nX4,256,0,stream>>>(hbott, x4);

  for (int li = 0; li < NLAY; ++li){
    for (int path = 0; path < 2; ++path){
      int T    = path ? SSEG : KSEG;
      int nseq = path ? NB*KSEG : NB*SSEG;
      int ntok = nseq*T;
      ln_kernel<<<cdiv(ntok,256),256,0,stream>>>(x4, ln_w + (li*2+path)*64, ln_b + (li*2+path)*64,
                                                 xnb, path, T, nseq);
      mamba_kernel<<<nseq*2,256,0,stream>>>(xnb, fbuf, w_in16, w_x16, w_o16,
                                            mm_conv_w, mm_conv_b, mm_dt_w, mm_dt_b,
                                            mm_Alog, mm_D, li, path, T, nseq);
      reduce_ir<<<NB*64,256,0,stream>>>(x4, fbuf, part, path, T, nseq);
      finalize_stats<<<NB,32,0,stream>>>(part, fin, (float)(N64*KS), 1e-8f);
      combine_kernel<<<nX4,256,0,stream>>>(x4, fbuf, gln4_w + (li*2+path)*64, gln4_b + (li*2+path)*64,
                                           fin, path, T, nseq);
    }
  }

  conv2d_kernel<<<nX4,256,0,stream>>>(x4, conv2d_w, conv2d_b, prelu_a, buf5);
  overadd_kernel<<<nL,256,0,stream>>>(buf5, hoa);
  mask1_kernel<<<nL,256,0,stream>>>(hoa, out_w, out_b, gate_w, gate_b, m1);
  mask2_kernel<<<nL,256,0,stream>>>(m1, end_w, mixw, mmk);
  dec_kernel<<<cdiv(NB*TLEN,256),256,0,stream>>>(mmk, dec_w, est);
}
